// Attention_6554120094191
// MI455X (gfx1250) — matrix operand order repaired and checked by reading, not compiled, not run on hardware
//
#include <hip/hip_runtime.h>

typedef __attribute__((ext_vector_type(16))) __bf16 v16bf;
typedef __attribute__((ext_vector_type(8)))  float  v8f;

#define S_LEN   2048
#define D_DIM   64
#define QBLK    128      // 8 waves * 16 query rows
#define KCHUNK  32
#define VT_PITCH 40      // 32 keys + 8 pad (keeps b64 fragment reads conflict-free)

// 1-instruction packed f32->bf16 (gfx1250 has full bf16 VALU): lo=cvt(a), hi=cvt(b)
__device__ __forceinline__ unsigned cvtpk_bf16(float a, float b) {
    unsigned r;
    asm("v_cvt_pk_bf16_f32 %0, %1, %2" : "=v"(r) : "v"(a), "v"(b));
    return r;
}
// cheap scalar round-to-nearest for the scattered V^T stores
__device__ __forceinline__ unsigned short f2bf_rn(float f) {
    unsigned u = __builtin_bit_cast(unsigned, f);
    return (unsigned short)((u + 0x8000u) >> 16);
}

union Frag16 {
    v16bf bf;
    uint4 q[2];
    uint2 d2[4];
    unsigned u[8];
};

// CDNA5 async copy: global -> LDS, 16 bytes per enabled lane, tracked by ASYNCcnt.
__device__ __forceinline__ void async_g2l_b128(unsigned ldsByteOff, const void* gptr) {
    asm volatile("global_load_async_to_lds_b128 %0, %1, off"
                 :: "v"(ldsByteOff), "v"((unsigned long long)gptr)
                 : "memory");
}
__device__ __forceinline__ void wait_asynccnt0() {
    asm volatile("s_wait_asynccnt 0x0" ::: "memory");
}

__global__ __launch_bounds__(256) void
attn_flash_wmma(const float* __restrict__ Q, const float* __restrict__ K,
                const float* __restrict__ V, float* __restrict__ O)
{
    __shared__ float          lKf[KCHUNK * D_DIM];   // f32 [key][d], filled by async DMA
    __shared__ unsigned short lKb[KCHUNK * D_DIM];   // bf16 [key][d], cooperative convert
    __shared__ unsigned short lVt[D_DIM * VT_PITCH]; // bf16 [d][key] (transposed)

    const int tid  = threadIdx.x;
    const int wave = tid >> 5;
    const int lane = tid & 31;
    const int l15  = lane & 15;
    const int half = (lane >> 4) & 1;     // 0: lanes 0-15, 1: lanes 16-31
    const bool lo  = (half == 0);

    const int bh   = blockIdx.y;          // 0..63  (b*H + h)
    const long headOff = (long)bh * S_LEN * D_DIM;
    const float* Qh = Q + headOff;
    const float* Kh = K + headOff;
    const float* Vh = V + headOff;
    float*       Oh = O + headOff;

    const int qRow = blockIdx.x * QBLK + wave * 16 + l15;

    // LDS byte offsets for this thread's two async K transfers (aperture rule:
    // low 32 bits of a generic shared pointer == LDS byte address).
    const unsigned ldsK0 = (unsigned)(unsigned long long)(const void*)&lKf[tid * 4];
    const unsigned ldsK1 = (unsigned)(unsigned long long)(const void*)&lKf[tid * 4 + 1024];

    // ---- Q^T B-operand fragments (K-dim = d). Lane n<16 holds query n,
    // d = f*32 + 0..7,16..23 ; lanes 16-31 hold d = f*32 + 8..15,24..31. ----
    Frag16 qf[2];
    {
        const float* qrow = Qh + (long)qRow * D_DIM;
#pragma unroll
        for (int f = 0; f < 2; ++f) {
            const int dbase = f * 32 + half * 8;
            float4 a = *(const float4*)(qrow + dbase + 0);
            float4 b = *(const float4*)(qrow + dbase + 4);
            float4 c = *(const float4*)(qrow + dbase + 16);
            float4 d = *(const float4*)(qrow + dbase + 20);
            qf[f].u[0] = cvtpk_bf16(a.x, a.y); qf[f].u[1] = cvtpk_bf16(a.z, a.w);
            qf[f].u[2] = cvtpk_bf16(b.x, b.y); qf[f].u[3] = cvtpk_bf16(b.z, b.w);
            qf[f].u[4] = cvtpk_bf16(c.x, c.y); qf[f].u[5] = cvtpk_bf16(c.z, c.w);
            qf[f].u[6] = cvtpk_bf16(d.x, d.y); qf[f].u[7] = cvtpk_bf16(d.z, d.w);
        }
    }

    v8f acc[4];
#pragma unroll
    for (int t = 0; t < 4; ++t) acc[t] = v8f{};
    float m = -3.0e38f, lsum = 0.0f;

    for (int kc = 0; kc < S_LEN; kc += KCHUNK) {
        // ---- K chunk: raw f32 via async DMA (no VGPR round-trip) ----
        async_g2l_b128(ldsK0, Kh + (long)kc * D_DIM + tid * 4);
        async_g2l_b128(ldsK1, Kh + (long)kc * D_DIM + tid * 4 + 1024);

        // ---- V chunk: manual transpose + f32->bf16 while the DMA runs ----
#pragma unroll
        for (int pass = 0; pass < 2; ++pass) {
            const int idx = tid * 4 + pass * 1024;  // element in [32 x 64]
            const int k  = idx >> 6;
            const int d0 = idx & 63;
            float4 vv = *(const float4*)(Vh + (long)(kc + k) * D_DIM + d0);
            lVt[(d0 + 0) * VT_PITCH + k] = f2bf_rn(vv.x);
            lVt[(d0 + 1) * VT_PITCH + k] = f2bf_rn(vv.y);
            lVt[(d0 + 2) * VT_PITCH + k] = f2bf_rn(vv.z);
            lVt[(d0 + 3) * VT_PITCH + k] = f2bf_rn(vv.w);
        }
        if (kc + KCHUNK < S_LEN) {   // prefetch next chunk -> global_prefetch_b8
            const long nrow = (long)(kc + KCHUNK + (tid >> 3)) * D_DIM;
            __builtin_prefetch(Kh + nrow, 0, 0);
            __builtin_prefetch(Vh + nrow, 0, 0);
        }
        wait_asynccnt0();
        __syncthreads();

        // ---- cooperative K f32 -> bf16 convert (once per chunk, not per wave) ----
        {
            const int idx = tid * 8;                 // 2048 elems / 256 threads
            float4 a = *(const float4*)&lKf[idx];
            float4 b = *(const float4*)&lKf[idx + 4];
            uint4 p;
            p.x = cvtpk_bf16(a.x, a.y); p.y = cvtpk_bf16(a.z, a.w);
            p.z = cvtpk_bf16(b.x, b.y); p.w = cvtpk_bf16(b.z, b.w);
            *(uint4*)&lKb[idx] = p;
        }
        __syncthreads();

        // ---- S^T tiles: ST = K_chunk * Q^T  (rows = keys, cols = queries) ----
        v8f st0 = v8f{}, st1 = v8f{};
        {
            Frag16 ka;
            const unsigned short* rp0 = &lKb[l15 * D_DIM];          // keys 0-15
            const unsigned short* rp1 = &lKb[(16 + l15) * D_DIM];   // keys 16-31
            ka.q[0] = *(const uint4*)(rp0 + (0  + half * 8));
            ka.q[1] = *(const uint4*)(rp0 + (16 + half * 8));
            st0 = __builtin_amdgcn_wmma_f32_16x16x32_bf16(false, ka.bf, false, qf[0].bf,
                                                          (short)0, st0, false, false);
            ka.q[0] = *(const uint4*)(rp0 + (32 + half * 8));
            ka.q[1] = *(const uint4*)(rp0 + (48 + half * 8));
            st0 = __builtin_amdgcn_wmma_f32_16x16x32_bf16(false, ka.bf, false, qf[1].bf,
                                                          (short)0, st0, false, false);
            ka.q[0] = *(const uint4*)(rp1 + (0  + half * 8));
            ka.q[1] = *(const uint4*)(rp1 + (16 + half * 8));
            st1 = __builtin_amdgcn_wmma_f32_16x16x32_bf16(false, ka.bf, false, qf[0].bf,
                                                          (short)0, st1, false, false);
            ka.q[0] = *(const uint4*)(rp1 + (32 + half * 8));
            ka.q[1] = *(const uint4*)(rp1 + (48 + half * 8));
            st1 = __builtin_amdgcn_wmma_f32_16x16x32_bf16(false, ka.bf, false, qf[1].bf,
                                                          (short)0, st1, false, false);
        }

        // ---- online softmax (per query == per lane-pair) ----
        float cmax = -3.0e38f;
#pragma unroll
        for (int v = 0; v < 8; ++v) cmax = fmaxf(cmax, fmaxf(st0[v], st1[v]));
        cmax = fmaxf(cmax, __shfl_xor(cmax, 16, 32));
        const float mnew = fmaxf(m, cmax);
        const float corr = __expf(m - mnew);
        m = mnew;

        float p0[8], p1[8], s = 0.0f;
#pragma unroll
        for (int v = 0; v < 8; ++v) {
            p0[v] = __expf(st0[v] - mnew);
            p1[v] = __expf(st1[v] - mnew);
            s += p0[v] + p1[v];
        }
        s += __shfl_xor(s, 16, 32);
        lsum = lsum * corr + s;
#pragma unroll
        for (int t = 0; t < 4; ++t)
#pragma unroll
            for (int v = 0; v < 8; ++v) acc[t][v] *= corr;

        // ---- build P^T B-operand fragment via cross-half shuffles ----
        // lane n<16 needs keys 0..15 of query n; lane n+16 needs keys 16..31.
        Frag16 pf;
#pragma unroll
        for (int w = 0; w < 8; ++w) {
            float x0, x1;
            if (w < 4) {                     // fragment K elems 2w, 2w+1  (keys 0-7 / 16-23)
                const float a0 = p0[2 * w], a1 = p0[2 * w + 1];
                const float b0 = __shfl_xor(p1[2 * w],     16, 32);
                const float b1 = __shfl_xor(p1[2 * w + 1], 16, 32);
                x0 = lo ? a0 : b0; x1 = lo ? a1 : b1;
            } else {                         // keys 8-15 / 24-31
                const int v = 2 * (w - 4);
                const float a0 = __shfl_xor(p0[v],     16, 32);
                const float a1 = __shfl_xor(p0[v + 1], 16, 32);
                const float b0 = p1[v], b1 = p1[v + 1];
                x0 = lo ? a0 : b0; x1 = lo ? a1 : b1;
            }
            pf.u[w] = cvtpk_bf16(x0, x1);
        }

        // ---- O^T += V^T * P^T : 4 M-tiles over d ----
#pragma unroll
        for (int t = 0; t < 4; ++t) {
            const unsigned short* rp = &lVt[(t * 16 + l15) * VT_PITCH];
            Frag16 va;
            va.d2[0] = *(const uint2*)(rp + half * 16 + 0);
            va.d2[1] = *(const uint2*)(rp + half * 16 + 4);
            va.d2[2] = *(const uint2*)(rp + half * 16 + 8);
            va.d2[3] = *(const uint2*)(rp + half * 16 + 12);
            acc[t] = __builtin_amdgcn_wmma_f32_16x16x32_bf16(false, va.bf, false, pf.bf,
                                                             (short)0, acc[t], false, false);
        }
        __syncthreads();
    }

    // ---- normalize and store: lane holds query q = l15, d = 16t + half*8 + v ----
    const float rl = 1.0f / lsum;
    float* orow = Oh + (long)qRow * D_DIM;
#pragma unroll
    for (int t = 0; t < 4; ++t) {
        float4 f0, f1;
        f0.x = acc[t][0] * rl; f0.y = acc[t][1] * rl;
        f0.z = acc[t][2] * rl; f0.w = acc[t][3] * rl;
        f1.x = acc[t][4] * rl; f1.y = acc[t][5] * rl;
        f1.z = acc[t][6] * rl; f1.w = acc[t][7] * rl;
        float* p = orow + t * 16 + half * 8;
        *(float4*)(p + 0) = f0;
        *(float4*)(p + 4) = f1;
    }
}

extern "C" void kernel_launch(void* const* d_in, const int* in_sizes, int n_in,
                              void* d_out, int out_size, void* d_ws, size_t ws_size,
                              hipStream_t stream) {
    const float* Q = (const float*)d_in[0];
    const float* K = (const float*)d_in[1];
    const float* V = (const float*)d_in[2];
    float* O = (float*)d_out;
    (void)in_sizes; (void)n_in; (void)out_size; (void)d_ws; (void)ws_size;
    dim3 grid(S_LEN / QBLK, 4 * 16);   // (q-blocks, B*H)
    attn_flash_wmma<<<grid, 256, 0, stream>>>(Q, K, V, O);
}
